// SelfAttention_5162550690439
// MI455X (gfx1250) — matrix operand order repaired and checked by reading, not compiled, not run on hardware
//
#include <hip/hip_runtime.h>
#include <math.h>

// ---------------------------------------------------------------------------
// Self-attention (B=4, S=2048, H=1024, fp32 ref) on MI455X / gfx1250.
// bf16 WMMA (v_wmma_f32_16x16x32_bf16) everywhere, f32 accumulation,
// flash-attention streaming softmax, V stored transposed so every B-matrix
// fragment load is a contiguous b128, and double-buffered
// GLOBAL_LOAD_ASYNC_TO_LDS_B128 staging (ASYNCcnt) for the GEMM weight tiles.
// ---------------------------------------------------------------------------

typedef __bf16 bf16;
typedef __attribute__((ext_vector_type(16))) __bf16 v16bf;
typedef __attribute__((ext_vector_type(8)))  __bf16 v8bf;
typedef __attribute__((ext_vector_type(4)))  __bf16 v4bf;
typedef __attribute__((ext_vector_type(8)))  float  v8f;

static constexpr int Bn = 4;
static constexpr int Sn = 2048;
static constexpr int Hn = 1024;
static constexpr int Mn = Bn * Sn;          // 8192 token rows
static constexpr float SCALE = 0.03125f;    // 1/sqrt(1024)

// --------------------------- device helpers --------------------------------

__device__ __forceinline__ v16bf cat16(v8bf lo, v8bf hi) {
    v16bf r;
#pragma unroll
    for (int i = 0; i < 8; ++i) { r[i] = lo[i]; r[i + 8] = hi[i]; }
    return r;
}

// Build a 16-half fragment register from two 16B chunks (global or LDS ptr).
__device__ __forceinline__ v16bf ld16(const bf16* p0, const bf16* p1) {
    return cat16(*(const v8bf*)p0, *(const v8bf*)p1);
}

__device__ __forceinline__ v8f wmma_bf16(v16bf a, v16bf b, v8f c) {
    // D = A(16x32) * B(32x16) + C(16x16 f32)
    return __builtin_amdgcn_wmma_f32_16x16x32_bf16(
        false, a, false, b, (short)0, c, false, false);
}

// CDNA5 async copy: DMA 16B from global directly into LDS, tracked by ASYNCcnt.
// ldsOff = low 32 bits of the generic LDS pointer (== LDS byte offset).
__device__ __forceinline__ void async_copy_b128(unsigned ldsOff, const bf16* gsrc) {
    asm volatile("global_load_async_to_lds_b128 %0, %1, off"
                 :: "v"(ldsOff), "v"((unsigned long long)(uintptr_t)gsrc)
                 : "memory");
}
__device__ __forceinline__ void wait_async0() {
    asm volatile("s_wait_asynccnt 0x0" ::: "memory");
}

// --------------------------- kernel 1: fp32 -> bf16 -------------------------

__global__ void k_cvt(const float* __restrict__ in, bf16* __restrict__ out, int n4) {
    int i = blockIdx.x * blockDim.x + threadIdx.x;
    int stride = gridDim.x * blockDim.x;
    for (; i < n4; i += stride) {
        float4 f = ((const float4*)in)[i];
        v4bf o;
        o[0] = (bf16)f.x; o[1] = (bf16)f.y; o[2] = (bf16)f.z; o[3] = (bf16)f.w;
        ((v4bf*)out)[i] = o;
    }
}

// --------------------------- kernel 2: QKV projection ------------------------
// Y[m,n] = sum_k Xb[m,k] * W[n,k] + bias[n]     (nn.Linear: x @ W^T + b)
// Block = 256 threads (8 waves). Wave w owns rows m0+16w..+15; block tile is
// 128(M) x 64(N). The 64x32 W tile is staged in LDS by double-buffered async
// DMA copies: while the waves run WMMAs on buffer `cur`, the async engine
// fills buffer `cur^1` with the next k-step's tile.
// vtrans==0: store bf16 Y[m][n] row-major ([B*S][H])     (Q, K)
// vtrans==1: store bf16 Y transposed as [B][H][S]        (V^T)

__global__ __launch_bounds__(256) void k_qkv(const bf16* __restrict__ Xb,
                                             const bf16* __restrict__ W,
                                             const float* __restrict__ bias,
                                             bf16* __restrict__ Out,
                                             int vtrans) {
    __shared__ __align__(16) bf16 sW[2][64][40];  // 2 x (64 rows x 32 k, +8 pad)

    const int tid  = threadIdx.x;
    const int wave = tid >> 5, lane = tid & 31;
    const int llo  = lane & 15, hi = lane >> 4;
    const int m0   = blockIdx.x * 128 + wave * 16;
    const int n0   = blockIdx.y * 64;

    v8f acc[4] = { {}, {}, {}, {} };

    const int wrow = tid >> 2;                    // 0..63
    const int wcol = (tid & 3) * 8;               // 0,8,16,24  (halves)

    // Each thread owns one 16B slice of the staged tile.
    const unsigned ldsOff0 = (unsigned)(uintptr_t)&sW[0][wrow][wcol];
    const unsigned ldsOff1 = (unsigned)(uintptr_t)&sW[1][wrow][wcol];
    const bf16* wsrc = &W[(size_t)(n0 + wrow) * Hn + wcol];

    async_copy_b128(ldsOff0, wsrc);               // prime tile for kk = 0
    wait_async0();
    __syncthreads();

    for (int kk = 0; kk < Hn; kk += 32) {
        const int cur = (kk >> 5) & 1;
        if (kk + 32 < Hn)                         // start DMA of next tile
            async_copy_b128(cur ? ldsOff0 : ldsOff1, wsrc + kk + 32);

        const bf16* xp = &Xb[(size_t)(m0 + llo) * Hn + kk + 8 * hi];
        v16bf A = ld16(xp, xp + 16);              // 16x32 A-fragment

#pragma unroll
        for (int t = 0; t < 4; ++t) {
            const bf16* wp = &sW[cur][t * 16 + llo][8 * hi];
            acc[t] = wmma_bf16(A, ld16(wp, wp + 16), acc[t]);
        }

        wait_async0();                            // next tile landed in LDS
        __syncthreads();
    }

    if (!vtrans) {
#pragma unroll
        for (int t = 0; t < 4; ++t) {
            const int col = n0 + t * 16 + llo;
            const float bv = bias[col];
#pragma unroll
            for (int r = 0; r < 8; ++r) {
                const int mrow = m0 + r + 8 * hi;
                Out[(size_t)mrow * Hn + col] = (bf16)(acc[t][r] + bv);
            }
        }
    } else {
        const int bb = m0 >> 11;                  // batch (128 | 2048)
        const int s0 = (m0 & (Sn - 1)) + 8 * hi;  // 8 consecutive s per store
#pragma unroll
        for (int t = 0; t < 4; ++t) {
            const int dv = n0 + t * 16 + llo;
            const float bv = bias[dv];
            v8bf o;
#pragma unroll
            for (int r = 0; r < 8; ++r) o[r] = (bf16)(acc[t][r] + bv);
            *(v8bf*)&Out[((size_t)bb * Hn + dv) * Sn + s0] = o;
        }
    }
}

// --------------------------- kernel 3: flash attention -----------------------
// One block per (batch, 16-row q tile). 8 waves:
//   scores: d=1024 contraction split across waves (128 each), partial 16x32
//           f32 tiles reduced through LDS, scaled by 1/sqrt(H).
//   softmax: 16 lanes do the streaming-max/exp/sum update, emit P (bf16) +
//            per-row correction factors.
//   P.V:    wave w owns output columns [128w,128w+128); 8 WMMAs per 32-key
//           chunk against V^T (contiguous fragment loads).

__global__ __launch_bounds__(256) void k_attn(const bf16* __restrict__ Qb,
                                              const bf16* __restrict__ Kb,
                                              const bf16* __restrict__ Vt,
                                              float* __restrict__ Out) {
    __shared__ float sPart[8][16][34];            // per-wave partial scores
    __shared__ float sS[16][32];                  // reduced, scaled scores
    __shared__ bf16  sP[16][40];                  // probabilities (padded)
    __shared__ float sM[16], sL[16], sCorr[16];

    const int tid  = threadIdx.x;
    const int wave = tid >> 5, lane = tid & 31;
    const int llo  = lane & 15, hi = lane >> 4;
    const int b    = blockIdx.x >> 7;
    const int q0   = (blockIdx.x & 127) * 16;

    // Preload this wave's Q d-slice: 4 A-fragments (16 x 32 each), reused for
    // every key chunk.
    v16bf AQ[4];
    const bf16* qrow = Qb + (size_t)(b * Sn + q0 + llo) * Hn + wave * 128;
#pragma unroll
    for (int i = 0; i < 4; ++i) {
        const bf16* qp = qrow + i * 32 + 8 * hi;
        AQ[i] = ld16(qp, qp + 16);
    }

    v8f acc[8];
#pragma unroll
    for (int nt = 0; nt < 8; ++nt) acc[nt] = v8f{};

    if (tid < 16) { sM[tid] = -3.0e38f; sL[tid] = 0.f; }
    __syncthreads();

    for (int kb = 0; kb < Sn; kb += 32) {
        // Prefetch next chunk's K/V cachelines (global_prefetch_b8).
        if (kb + 32 < Sn) {
            __builtin_prefetch(Kb + (size_t)(b * Sn + kb + 32 + llo) * Hn
                                  + wave * 128 + 16 * hi, 0, 0);
            __builtin_prefetch(Vt + ((size_t)b * Hn + wave * 128 + llo) * Sn
                                  + kb + 32 + 16 * hi, 0, 0);
        }

        // ---- partial scores over this wave's 128-wide d slice -------------
        v8f s0 = {}, s1 = {};
#pragma unroll
        for (int i = 0; i < 4; ++i) {
            const int d = wave * 128 + i * 32;
            const bf16* kp0 = Kb + (size_t)(b * Sn + kb + llo) * Hn + d + 8 * hi;
            const bf16* kp1 = Kb + (size_t)(b * Sn + kb + 16 + llo) * Hn + d + 8 * hi;
            s0 = wmma_bf16(AQ[i], ld16(kp0, kp0 + 16), s0);
            s1 = wmma_bf16(AQ[i], ld16(kp1, kp1 + 16), s1);
        }
#pragma unroll
        for (int r = 0; r < 8; ++r) {
            sPart[wave][r + 8 * hi][llo]      = s0[r];
            sPart[wave][r + 8 * hi][16 + llo] = s1[r];
        }
        __syncthreads();

        // ---- cross-wave reduction + scale ---------------------------------
#pragma unroll
        for (int e = tid; e < 512; e += 256) {
            const int m = e >> 5, n = e & 31;
            float v = 0.f;
#pragma unroll
            for (int w = 0; w < 8; ++w) v += sPart[w][m][n];
            sS[m][n] = v * SCALE;
        }
        __syncthreads();

        // ---- streaming softmax (one lane per q row) -----------------------
        if (tid < 16) {
            const float mOld = sM[tid];
            float mNew = mOld;
            float row[32];
#pragma unroll
            for (int j = 0; j < 32; ++j) {
                row[j] = sS[tid][j];
                mNew = fmaxf(mNew, row[j]);
            }
            const float corr = (mOld <= -3.0e38f) ? 0.f : __expf(mOld - mNew);
            float sum = 0.f;
#pragma unroll
            for (int j = 0; j < 32; ++j) {
                const float p = __expf(row[j] - mNew);
                sum += p;
                sP[tid][j] = (bf16)p;
            }
            sCorr[tid] = corr;
            sM[tid]    = mNew;
            sL[tid]    = sL[tid] * corr + sum;
        }
        __syncthreads();

        // ---- rescale accumulators, then P(16x32) @ V(32x128 slice) --------
        v16bf AP = ld16(&sP[llo][16 * hi], &sP[llo][16 * hi + 8]);
        float c[8];
#pragma unroll
        for (int r = 0; r < 8; ++r) c[r] = sCorr[r + 8 * hi];

#pragma unroll
        for (int nt = 0; nt < 8; ++nt) {
#pragma unroll
            for (int r = 0; r < 8; ++r) acc[nt][r] *= c[r];
            const bf16* vp = Vt + ((size_t)b * Hn + wave * 128 + nt * 16 + llo) * Sn
                             + kb + 16 * hi;
            acc[nt] = wmma_bf16(AP, ld16(vp, vp + 8), acc[nt]);
        }
        // next-iter barrier (after sPart store) guards all LDS reuse
    }

    // ---- finalize: divide by row sums, store fp32 --------------------------
    float linv[8];
#pragma unroll
    for (int r = 0; r < 8; ++r) linv[r] = 1.0f / sL[r + 8 * hi];
#pragma unroll
    for (int nt = 0; nt < 8; ++nt) {
        const int dv = wave * 128 + nt * 16 + llo;
#pragma unroll
        for (int r = 0; r < 8; ++r) {
            const int mrow = q0 + r + 8 * hi;
            Out[(size_t)(b * Sn + mrow) * Hn + dv] = acc[nt][r] * linv[r];
        }
    }
}

// --------------------------- launch ------------------------------------------

extern "C" void kernel_launch(void* const* d_in, const int* in_sizes, int n_in,
                              void* d_out, int out_size, void* d_ws, size_t ws_size,
                              hipStream_t stream) {
    (void)in_sizes; (void)n_in; (void)out_size; (void)ws_size;

    const float* x  = (const float*)d_in[0];
    const float* Wq = (const float*)d_in[1];
    const float* bq = (const float*)d_in[2];
    const float* Wk = (const float*)d_in[3];
    const float* bk = (const float*)d_in[4];
    const float* Wv = (const float*)d_in[5];
    const float* bv = (const float*)d_in[6];
    float* out = (float*)d_out;

    // Workspace layout (bf16): xb 16MiB | Wq/Wk/Wv 2MiB each | Q 16 | K 16 | Vt 16
    char* ws = (char*)d_ws;
    bf16* xb  = (bf16*)(ws);
    bf16* Wqb = (bf16*)(ws + (size_t)16 * 1024 * 1024);
    bf16* Wkb = (bf16*)(ws + (size_t)18 * 1024 * 1024);
    bf16* Wvb = (bf16*)(ws + (size_t)20 * 1024 * 1024);
    bf16* Qb  = (bf16*)(ws + (size_t)22 * 1024 * 1024);
    bf16* Kb  = (bf16*)(ws + (size_t)38 * 1024 * 1024);
    bf16* Vt  = (bf16*)(ws + (size_t)54 * 1024 * 1024);

    k_cvt<<<2048, 256, 0, stream>>>(x,  xb,  (Mn * Hn) / 4);
    k_cvt<<<512,  256, 0, stream>>>(Wq, Wqb, (Hn * Hn) / 4);
    k_cvt<<<512,  256, 0, stream>>>(Wk, Wkb, (Hn * Hn) / 4);
    k_cvt<<<512,  256, 0, stream>>>(Wv, Wvb, (Hn * Hn) / 4);

    dim3 g2(Mn / 128, Hn / 64);                  // (64, 16)
    k_qkv<<<g2, 256, 0, stream>>>(xb, Wqb, bq, Qb, 0);
    k_qkv<<<g2, 256, 0, stream>>>(xb, Wkb, bk, Kb, 0);
    k_qkv<<<g2, 256, 0, stream>>>(xb, Wvb, bv, Vt, 1);   // V stored [B][H][S]

    k_attn<<<Bn * (Sn / 16), 256, 0, stream>>>(Qb, Kb, Vt, out);
}
